// ExtSummModel_39548058862133
// MI455X (gfx1250) — matrix operand order repaired and checked by reading, not compiled, not run on hardware
//
#include <hip/hip_runtime.h>
#include <hip/hip_bf16.h>

typedef __attribute__((ext_vector_type(16))) __bf16 bf16x16;
typedef __attribute__((ext_vector_type(8)))  __bf16 bf16x8;
typedef __attribute__((ext_vector_type(8)))  float  f32x8;

// Model dims
static constexpr int Bn = 32, Sn = 512, Hn = 128, En = 300, EP = 320, Tn = 16, Wn = 20;
static constexpr int BS = Bn * Sn;          // 16384 rows
static constexpr int G3H = 3 * Hn;          // 384
static constexpr int H4 = 4 * Hn;           // 512

__device__ __forceinline__ float sigmoidf_(float x) { return 1.f / (1.f + __expf(-x)); }

// ---------------------------------------------------------------------------
// 1) Sentence encoder: mean of 20 word embeddings -> bf16, K-padded 300->320
// ---------------------------------------------------------------------------
__global__ void __launch_bounds__(128)
embed_mean_kernel(const int* __restrict__ word_ids, const float* __restrict__ emb,
                  __bf16* __restrict__ xbf)
{
  __shared__ int ids[Wn];
  const int bs = blockIdx.x;                       // b*S + s
  if (threadIdx.x < Wn) ids[threadIdx.x] = word_ids[(size_t)bs * Wn + threadIdx.x];
  __syncthreads();
  for (int e = threadIdx.x; e < EP; e += 128) {
    float acc = 0.f;
    if (e < En) {
#pragma unroll
      for (int w = 0; w < Wn; ++w) acc += emb[(size_t)ids[w] * En + e];
      acc *= (1.f / (float)Wn);
    }
    xbf[(size_t)bs * EP + e] = (__bf16)acc;
  }
}

// ---------------------------------------------------------------------------
// 2) Weight conversion: fp32 -> bf16, row-major [R,Cp] (Bt layout), zero-pad,
//    optional transpose (src [C,R] -> dst[r,c] = src[c,r]).
// ---------------------------------------------------------------------------
__global__ void convert_pad_kernel(const float* __restrict__ src, __bf16* __restrict__ dst,
                                   int R, int C, int Cp, int transpose)
{
  const int idx = blockIdx.x * blockDim.x + threadIdx.x;
  if (idx >= R * Cp) return;
  const int r = idx / Cp, c = idx - r * Cp;
  float v = 0.f;
  if (c < C) v = transpose ? src[(size_t)c * R + r] : src[(size_t)r * C + c];
  dst[idx] = (__bf16)v;
}

// ---------------------------------------------------------------------------
// 3) Register-blocked bf16 WMMA GEMM: C[M,N] = act(A[M,K] @ Bt[N,K]^T + bias)
//    Each wave computes a 64x64 output block: 4 A-frags x 4 B-frags reused
//    into 16 WMMAs per 32-wide K-step (1 wmma : 1 b128-load).
//    A,Bt bf16 row-major; K % 32 == 0; M % 64 == 0; N % 64 == 0.
//    act: 0=none, 1=relu, 2=tanh.  8 waves/block.
// ---------------------------------------------------------------------------
__global__ void __launch_bounds__(256)
gemm_bf16_kernel(const __bf16* __restrict__ A, const __bf16* __restrict__ Bt,
                 const float* __restrict__ bias, float* __restrict__ C,
                 int M, int N, int K, int act)
{
  const int lane = threadIdx.x & 31;
  const int wave = threadIdx.x >> 5;
  const int job  = blockIdx.x * 8 + wave;
  const int nStr = N >> 6;              // 64-wide strips
  const int jM   = job / nStr;
  const int jN   = job - jM * nStr;
  if (jM * 64 >= M) return;

  const int mBase = jM * 64 + (lane & 15);
  const int nBase = jN * 64 + (lane & 15);
  const int aOff  = (lane >> 4) << 4;   // A: lanes>=16 start at K+16
  const int bOff  = (lane >> 4) << 4;   // B: lanes>=16 start at K+16

  f32x8 acc[4][4];
#pragma unroll
  for (int i = 0; i < 4; ++i)
#pragma unroll
    for (int j = 0; j < 4; ++j)
      acc[i][j] = (f32x8){0.f, 0.f, 0.f, 0.f, 0.f, 0.f, 0.f, 0.f};

  for (int k0 = 0; k0 < K; k0 += 32) {
    bf16x16 af[4], bfr[4];
#pragma unroll
    for (int i = 0; i < 4; ++i) {
      const __bf16* ap = A + (size_t)(mBase + 16 * i) * K + k0 + aOff;
      bf16x8 lo = *(const bf16x8*)ap;          // K = base+0..7
      bf16x8 hi = *(const bf16x8*)(ap + 8);    // K = base+8..15
      af[i] = __builtin_shufflevector(lo, hi, 0,1,2,3,4,5,6,7,8,9,10,11,12,13,14,15);
      bfr[i] = *(const bf16x16*)(Bt + (size_t)(nBase + 16 * i) * K + k0 + bOff);
    }
#pragma unroll
    for (int i = 0; i < 4; ++i)
#pragma unroll
      for (int j = 0; j < 4; ++j)
        acc[i][j] = __builtin_amdgcn_wmma_f32_16x16x32_bf16(false, af[i], false, bfr[j],
                                                            (short)0, acc[i][j], false, false);
  }

  const int rBase = jM * 64 + ((lane >> 4) << 3);
#pragma unroll
  for (int j = 0; j < 4; ++j) {
    const int n = nBase + 16 * j;
    const float bz = bias ? bias[n] : 0.f;
#pragma unroll
    for (int i = 0; i < 4; ++i) {
      const int mr = rBase + 16 * i;
#pragma unroll
      for (int r = 0; r < 8; ++r) {
        float v = acc[i][j][r] + bz;
        if (act == 1) v = fmaxf(v, 0.f);
        else if (act == 2) v = tanhf(v);
        C[(size_t)(mr + r) * N + n] = v;
      }
    }
  }
}

// ---------------------------------------------------------------------------
// 4) Persistent biGRU scan: 2 blocks (dir), 256 threads (8 waves).
//    Per step: gh[32,384] = h[32,128] @ Whh^T via 48 WMMA tiles (6/wave),
//    then fused gate math. Hidden fp32 lives in registers (16 elems/thread),
//    bf16 copy in LDS feeds the next step's WMMA A-operand.
// ---------------------------------------------------------------------------
__global__ void __launch_bounds__(256)
gru_scan_kernel(const float* __restrict__ GXf, const float* __restrict__ GXb,
                const __bf16* __restrict__ WhhTf, const __bf16* __restrict__ WhhTb,
                const float* __restrict__ bhhF, const float* __restrict__ bhhB,
                float* __restrict__ sent_rep, float* __restrict__ lastF,
                float* __restrict__ lastB)
{
  const int dir = blockIdx.x;
  const float*  GX   = dir ? GXb   : GXf;
  const __bf16* Whh  = dir ? WhhTb : WhhTf;
  const float*  bhh  = dir ? bhhB  : bhhF;
  float*        lastO = dir ? lastB : lastF;

  __shared__ __bf16 h_bf[Bn * Hn];      // 8 KB
  __shared__ float  gh[Bn * G3H];       // 48 KB
  __shared__ float  bhh_s[G3H];         // 1.5 KB

  const int tid = threadIdx.x;
  for (int i = tid; i < G3H; i += 256) bhh_s[i] = bhh[i];
  for (int i = tid; i < Bn * Hn; i += 256) h_bf[i] = (__bf16)0.f;
  float hreg[16];
#pragma unroll
  for (int c = 0; c < 16; ++c) hreg[c] = 0.f;
  __syncthreads();

  const int lane = tid & 31;
  const int wave = tid >> 5;
  const int mloc = lane & 15;
  const int aOff = (lane >> 4) << 4;
  const int bOff = (lane >> 4) << 4;

  for (int t = 0; t < Sn; ++t) {
    const int s = dir ? (Sn - 1 - t) : t;

    // Phase A: gh = h @ Whh^T   (M=32, N=384, K=128)
#pragma unroll
    for (int i = 0; i < 6; ++i) {
      const int tile  = wave * 6 + i;
      const int tileM = tile / 24;
      const int tileN = tile - tileM * 24;
      const int m = tileM * 16 + mloc;
      const int n = tileN * 16 + mloc;
      f32x8 acc = {0.f, 0.f, 0.f, 0.f, 0.f, 0.f, 0.f, 0.f};
      const __bf16* ap = h_bf + m * Hn + aOff;
      const __bf16* bp = Whh + (size_t)n * Hn + bOff;
#pragma unroll
      for (int k0 = 0; k0 < Hn; k0 += 32) {
        bf16x8 alo = *(const bf16x8*)(ap + k0);
        bf16x8 ahi = *(const bf16x8*)(ap + k0 + 8);
        bf16x16 af = __builtin_shufflevector(alo, ahi, 0,1,2,3,4,5,6,7,8,9,10,11,12,13,14,15);
        bf16x16 bfrag = *(const bf16x16*)(bp + k0);
        acc = __builtin_amdgcn_wmma_f32_16x16x32_bf16(false, af, false, bfrag,
                                                      (short)0, acc, false, false);
      }
      const int mrow = tileM * 16 + ((lane >> 4) << 3);
#pragma unroll
      for (int r = 0; r < 8; ++r)
        gh[(mrow + r) * G3H + n] = acc[r];
    }
    __syncthreads();

    // Phase B: fused GRU gates (PyTorch order r,z,n) + hidden update
#pragma unroll
    for (int c = 0; c < 16; ++c) {
      const int i = c * 256 + tid;       // i == b*128 + j
      const int b = i >> 7;
      const int j = i & 127;
      const size_t row = ((size_t)b * Sn + s) * G3H;
      const float ghr = gh[b * G3H + j]            + bhh_s[j];
      const float ghz = gh[b * G3H + j + Hn]       + bhh_s[j + Hn];
      const float ghn = gh[b * G3H + j + 2 * Hn]   + bhh_s[j + 2 * Hn];
      const float r_ = sigmoidf_(GX[row + j] + ghr);
      const float z_ = sigmoidf_(GX[row + j + Hn] + ghz);
      const float n_ = tanhf(GX[row + j + 2 * Hn] + r_ * ghn);
      const float h2 = (1.f - z_) * n_ + z_ * hreg[c];
      hreg[c] = h2;
      h_bf[i] = (__bf16)h2;
      sent_rep[((size_t)b * Sn + s) * (2 * Hn) + dir * Hn + j] = h2;
    }
    __syncthreads();
  }
#pragma unroll
  for (int c = 0; c < 16; ++c) lastO[c * 256 + tid] = hreg[c];
}

// ---------------------------------------------------------------------------
// 5) doc_vec with the faithful torch .view(B,2H) quirk on [2,B,H]
// ---------------------------------------------------------------------------
__global__ void docvec_kernel(const float* __restrict__ lastF, const float* __restrict__ lastB,
                              float* __restrict__ doc_vec)
{
  const int idx = blockIdx.x * blockDim.x + threadIdx.x;
  if (idx >= Bn * 2 * Hn) return;
  const int b = idx >> 8, j = idx & 255;
  float v;
  if (b < 16) v = lastF[(2 * b + (j >> 7)) * Hn + (j & 127)];
  else        v = lastB[(2 * b - 32 + (j >> 7)) * Hn + (j & 127)];
  doc_vec[idx] = v;
}

// ---------------------------------------------------------------------------
// 6) Topic extraction (boundary diffs) + build bf16 concat inputs for attention
// ---------------------------------------------------------------------------
__global__ void __launch_bounds__(256)
build_cat_kernel(const float* __restrict__ sent_rep, const int* __restrict__ tse,
                 const float* __restrict__ doc_vec, float* __restrict__ topic_rep,
                 __bf16* __restrict__ cat_ds, __bf16* __restrict__ cat_ts)
{
  const int bs = blockIdx.x;
  const int b = bs >> 9;
  const int s = bs & (Sn - 1);
  const int pos = s + 1;                      // 1-based sentence id
  int seg = -1;
#pragma unroll
  for (int t = 0; t < Tn; ++t) seg += (tse[(b * Tn + t) * 2] <= pos) ? 1 : 0;
  const int st = tse[(b * Tn + seg) * 2];
  const int en = tse[(b * Tn + seg) * 2 + 1];

  const int j = threadIdx.x;                  // 0..255
  const size_t base = (size_t)bs * (2 * Hn);
  const float sv = sent_rep[base + j];
  float tv;
  if (j < Hn) {                               // forward diff: P[en] - P[st-1]
    const float a = sent_rep[((size_t)b * Sn + (en - 1)) * (2 * Hn) + j];
    const float c = (st >= 2) ? sent_rep[((size_t)b * Sn + (st - 2)) * (2 * Hn) + j] : 0.f;
    tv = a - c;
  } else {                                    // backward diff: P[st] - P[en+1]
    const float a = sent_rep[((size_t)b * Sn + (st - 1)) * (2 * Hn) + j];
    const float c = (en + 1 <= Sn) ? sent_rep[((size_t)b * Sn + en) * (2 * Hn) + j] : 0.f;
    tv = a - c;
  }
  const float dv = doc_vec[b * (2 * Hn) + j];
  topic_rep[base + j] = tv;
  const size_t cb = (size_t)bs * H4;
  cat_ds[cb + j]            = (__bf16)dv;
  cat_ds[cb + 2 * Hn + j]   = (__bf16)sv;
  cat_ts[cb + j]            = (__bf16)tv;
  cat_ts[cb + 2 * Hn + j]   = (__bf16)sv;
}

// ---------------------------------------------------------------------------
// 7) Row-wise dot with a vector (scores = tmp @ v_att; logits = h1 @ W2^T + b2)
// ---------------------------------------------------------------------------
__global__ void __launch_bounds__(256)
rowdot_kernel(const float* __restrict__ A, const float* __restrict__ v,
              const float* __restrict__ bias, float* __restrict__ out, int M, int K)
{
  const int lane = threadIdx.x & 31;
  const int row  = blockIdx.x * 8 + (threadIdx.x >> 5);
  if (row >= M) return;
  float sum = 0.f;
  for (int k = lane; k < K; k += 32) sum += A[(size_t)row * K + k] * v[k];
#pragma unroll
  for (int off = 16; off > 0; off >>= 1) sum += __shfl_xor(sum, off, 32);
  if (lane == 0) out[row] = sum + (bias ? bias[0] : 0.f);
}

// ---------------------------------------------------------------------------
// 8) Gated context + build bf16 decoder input [sent_rep, context]
// ---------------------------------------------------------------------------
__global__ void __launch_bounds__(256)
context_kernel(const float* __restrict__ sent_rep, const float* __restrict__ topic_rep,
               const float* __restrict__ doc_vec, const float* __restrict__ dsS,
               const float* __restrict__ tsS, __bf16* __restrict__ inp_bf)
{
  const int bs = blockIdx.x;
  const int b = bs >> 9;
  const int j = threadIdx.x;
  const float d = dsS[bs], t = tsS[bs];
  const float inv = 1.f / (d + t);
  const size_t base = (size_t)bs * (2 * Hn);
  const float ctx = (d * inv) * doc_vec[b * (2 * Hn) + j] + (t * inv) * topic_rep[base + j];
  const size_t cb = (size_t)bs * H4;
  inp_bf[cb + j]          = (__bf16)sent_rep[base + j];
  inp_bf[cb + 2 * Hn + j] = (__bf16)ctx;
}

// ---------------------------------------------------------------------------
extern "C" void kernel_launch(void* const* d_in, const int* in_sizes, int n_in,
                              void* d_out, int out_size, void* d_ws, size_t ws_size,
                              hipStream_t stream)
{
  const int*   word_ids = (const int*)d_in[0];
  const int*   tse      = (const int*)d_in[1];
  const float* emb      = (const float*)d_in[2];
  const float* Wih_f = (const float*)d_in[3];
  const float* Whh_f = (const float*)d_in[4];
  const float* bih_f = (const float*)d_in[5];
  const float* bhh_f = (const float*)d_in[6];
  const float* Wih_b = (const float*)d_in[7];
  const float* Whh_b = (const float*)d_in[8];
  const float* bih_b = (const float*)d_in[9];
  const float* bhh_b = (const float*)d_in[10];
  const float* W_att = (const float*)d_in[11];
  const float* v_att = (const float*)d_in[12];
  const float* W1 = (const float*)d_in[13];
  const float* b1 = (const float*)d_in[14];
  const float* W2 = (const float*)d_in[15];
  const float* b2 = (const float*)d_in[16];
  float* out = (float*)d_out;

  char* ws = (char*)d_ws;
  size_t cur = 0;
  auto alloc = [&](size_t bytes) -> void* {
    void* p = ws + cur;
    cur = (cur + bytes + 255) & ~(size_t)255;
    return p;
  };

  __bf16* xbf     = (__bf16*)alloc((size_t)BS * EP * 2);
  __bf16* wihf_bf = (__bf16*)alloc((size_t)G3H * EP * 2);
  __bf16* wihb_bf = (__bf16*)alloc((size_t)G3H * EP * 2);
  __bf16* whhf_bf = (__bf16*)alloc((size_t)G3H * Hn * 2);
  __bf16* whhb_bf = (__bf16*)alloc((size_t)G3H * Hn * 2);
  __bf16* watt_bf = (__bf16*)alloc((size_t)H4 * H4 * 2);
  __bf16* w1_bf   = (__bf16*)alloc((size_t)Hn * H4 * 2);
  float*  GXf     = (float*)alloc((size_t)BS * G3H * 4);   // GXf..GXb contiguous;
  float*  GXb     = (float*)alloc((size_t)BS * G3H * 4);   // tmp aliases after GRU
  float*  sent_rep= (float*)alloc((size_t)BS * 2 * Hn * 4);
  float*  last_f  = (float*)alloc((size_t)Bn * Hn * 4);
  float*  last_b  = (float*)alloc((size_t)Bn * Hn * 4);
  float*  doc_vec = (float*)alloc((size_t)Bn * 2 * Hn * 4);
  float*  topic_rep = (float*)alloc((size_t)BS * 2 * Hn * 4);
  __bf16* cat_ds  = (__bf16*)alloc((size_t)BS * H4 * 2);
  __bf16* cat_ts  = (__bf16*)alloc((size_t)BS * H4 * 2);
  float*  ds_sc   = (float*)alloc((size_t)BS * 4);
  float*  ts_sc   = (float*)alloc((size_t)BS * 4);
  __bf16* inp_bf  = (__bf16*)alloc((size_t)BS * H4 * 2);
  float*  h1      = (float*)alloc((size_t)BS * Hn * 4);
  float*  tmp     = GXf;   // 33.5 MB alias over 50.3 MB dead GX region

  // 1) sentence means (bf16, padded)
  embed_mean_kernel<<<BS, 128, 0, stream>>>(word_ids, emb, xbf);

  // 2) weight conversion to Bt bf16 layouts
  auto cvt = [&](const float* src, __bf16* dst, int R, int C, int Cp, int tr) {
    int n = R * Cp;
    convert_pad_kernel<<<(n + 255) / 256, 256, 0, stream>>>(src, dst, R, C, Cp, tr);
  };
  cvt(Wih_f, wihf_bf, G3H, En, EP, 0);   // Bt = Wih [384,320]
  cvt(Wih_b, wihb_bf, G3H, En, EP, 0);
  cvt(Whh_f, whhf_bf, G3H, Hn, Hn, 0);   // Bt = Whh [384,128]
  cvt(Whh_b, whhb_bf, G3H, Hn, Hn, 0);
  cvt(W_att, watt_bf, H4, H4, H4, 1);    // Bt = W_att^T [512,512]
  cvt(W1,    w1_bf,   Hn, H4, H4, 0);    // Bt = W1 [128,512]

  // 3) batched input projections GX = x @ Wih^T + bih  (bih folded as GEMM bias)
  {
    int jobs = (BS / 64) * (G3H / 64);   // 256 * 6 = 1536
    gemm_bf16_kernel<<<(jobs + 7) / 8, 256, 0, stream>>>(xbf, wihf_bf, bih_f, GXf, BS, G3H, EP, 0);
    gemm_bf16_kernel<<<(jobs + 7) / 8, 256, 0, stream>>>(xbf, wihb_bf, bih_b, GXb, BS, G3H, EP, 0);
  }

  // 4) persistent biGRU scan (one block per direction)
  gru_scan_kernel<<<2, 256, 0, stream>>>(GXf, GXb, whhf_bf, whhb_bf, bhh_f, bhh_b,
                                         sent_rep, last_f, last_b);

  // 5) doc vector (torch view quirk)
  docvec_kernel<<<(Bn * 2 * Hn + 255) / 256, 256, 0, stream>>>(last_f, last_b, doc_vec);

  // 6) topic segments + concat inputs
  build_cat_kernel<<<BS, 256, 0, stream>>>(sent_rep, tse, doc_vec, topic_rep, cat_ds, cat_ts);

  // 7) attention scores: tanh(cat @ W_att) @ v_att
  {
    int jobs = (BS / 64) * (H4 / 64);    // 256 * 8 = 2048
    gemm_bf16_kernel<<<(jobs + 7) / 8, 256, 0, stream>>>(cat_ds, watt_bf, nullptr, tmp, BS, H4, H4, 2);
    rowdot_kernel<<<BS / 8, 256, 0, stream>>>(tmp, v_att, nullptr, ds_sc, BS, H4);
    gemm_bf16_kernel<<<(jobs + 7) / 8, 256, 0, stream>>>(cat_ts, watt_bf, nullptr, tmp, BS, H4, H4, 2);
    rowdot_kernel<<<BS / 8, 256, 0, stream>>>(tmp, v_att, nullptr, ts_sc, BS, H4);
  }

  // 8) gated context + decoder input
  context_kernel<<<BS, 256, 0, stream>>>(sent_rep, topic_rep, doc_vec, ds_sc, ts_sc, inp_bf);

  // 9) h1 = relu(inp @ W1^T + b1); logits = h1 @ W2^T + b2
  {
    int jobs = (BS / 64) * (Hn / 64);    // 256 * 2 = 512
    gemm_bf16_kernel<<<(jobs + 7) / 8, 256, 0, stream>>>(inp_bf, w1_bf, b1, h1, BS, Hn, H4, 1);
    rowdot_kernel<<<BS / 8, 256, 0, stream>>>(h1, W2, b2, out, BS, Hn);
  }
  (void)in_sizes; (void)n_in; (void)out_size; (void)ws_size;
}
